// GAT_PointGeo_68221260530301
// MI455X (gfx1250) — matrix operand order repaired and checked by reading, not compiled, not run on hardware
//
#include <hip/hip_runtime.h>
#include <stdint.h>

#define N_NODES 10000
#define N_EDGES 160000
#define DIM     512
#define KPAIRS  256   // DIM/2 packed bf16 pairs per row

typedef __attribute__((ext_vector_type(16))) __bf16 v16bf;
typedef __attribute__((ext_vector_type(8)))  float  v8f;

union Frag16 { uint4 q[2]; v16bf h; };

__device__ __forceinline__ uint32_t pack_bf16(float lo, float hi) {
  uint32_t ul = __float_as_uint(lo);
  uint32_t uh = __float_as_uint(hi);
  ul = (ul + 0x7FFFu + ((ul >> 16) & 1u)) >> 16;   // round-to-nearest-even
  uh = (uh + 0x7FFFu + ((uh >> 16) & 1u)) >> 16;
  return (ul & 0xFFFFu) | (uh << 16);
}

// ---- prep: x (f32, row-major [N,D]) -> packed bf16 pairs -------------------
__global__ void k_cvt_x(const float* __restrict__ x, uint32_t* __restrict__ xbf) {
  int i = blockIdx.x * blockDim.x + threadIdx.x;
  if (i >= N_NODES * KPAIRS) return;
  float2 f = ((const float2*)x)[i];
  xbf[i] = pack_bf16(f.x, f.y);
}

// ---- prep: W (f32, row-major [D,D]) -> B-fragment layout Wpk[n*256 + k/2] --
__global__ void k_pack_w(const float* __restrict__ W, uint32_t* __restrict__ Wpk) {
  int i = blockIdx.x * blockDim.x + threadIdx.x;
  if (i >= DIM * KPAIRS) return;
  int n  = i >> 8;
  int kk = i & 255;
  float lo = W[(2 * kk)     * DIM + n];
  float hi = W[(2 * kk + 1) * DIM + n];
  Wpk[n * KPAIRS + kk] = pack_bf16(lo, hi);
}

// ---- fused dual GEMM: x_src = x@Ws + bs ; x_dst = x@Wd + bd (bf16 WMMA) ----
__global__ void __launch_bounds__(256) k_gemm_wmma(
    const uint32_t* __restrict__ xbf,
    const uint32_t* __restrict__ Wspk, const uint32_t* __restrict__ Wdpk,
    const float* __restrict__ bs, const float* __restrict__ bd,
    float* __restrict__ xsrc, float* __restrict__ xdst) {
  const int wave = threadIdx.x >> 5;
  const int lane = threadIdx.x & 31;
  const int tile = blockIdx.x * 8 + wave;             // 625*32 = 20000 tiles
  const int m0 = (tile >> 5) << 4;
  const int n0 = (tile & 31) << 4;
  const int lm = lane & 15;
  const int hi = lane >> 4;
  const int n  = n0 + lm;

  v8f cs = {};
  v8f cd = {};
  const uint32_t* arow  = xbf  + (m0 + lm) * KPAIRS + hi * 8;   // A frag (contiguous k = 16h + s)
  const uint32_t* bsrow = Wspk + n * KPAIRS + hi * 8;           // B frag
  const uint32_t* bdrow = Wdpk + n * KPAIRS + hi * 8;

  for (int k0 = 0; k0 < DIM; k0 += 32) {
    const int kk = k0 >> 1;
    Frag16 A, Bs, Bd;
    const uint4* pa = (const uint4*)(arow + kk);
    A.q[0] = pa[0];
    A.q[1] = pa[1];
    const uint4* pb = (const uint4*)(bsrow + kk);
    Bs.q[0] = pb[0]; Bs.q[1] = pb[1];
    const uint4* pc = (const uint4*)(bdrow + kk);
    Bd.q[0] = pc[0]; Bd.q[1] = pc[1];
    cs = __builtin_amdgcn_wmma_f32_16x16x32_bf16(false, A.h, false, Bs.h,
                                                 (short)0, cs, false, false);
    cd = __builtin_amdgcn_wmma_f32_16x16x32_bf16(false, A.h, false, Bd.h,
                                                 (short)0, cd, false, false);
  }
  const float bsn = bs[n];
  const float bdn = bd[n];
#pragma unroll
  for (int r = 0; r < 8; ++r) {
    int m = m0 + r + hi * 8;
    xsrc[m * DIM + n] = cs[r] + bsn;
    xdst[m * DIM + n] = cd[r] + bdn;
  }
}

// ---- counting sort of edges by destination ---------------------------------
__global__ void k_zero_cnt(int* __restrict__ cnt) {
  int i = blockIdx.x * blockDim.x + threadIdx.x;
  if (i < N_NODES) cnt[i] = 0;
}

__global__ void k_hist(const int* __restrict__ ei, int* __restrict__ cnt) {
  int e = blockIdx.x * blockDim.x + threadIdx.x;
  if (e >= N_EDGES) return;
  atomicAdd(&cnt[ei[N_EDGES + e]], 1);
}

#define SCAN_T 1024
#define SCAN_ITEMS 10          // 1024*10 >= N_NODES
__global__ void __launch_bounds__(SCAN_T) k_scan(const int* __restrict__ cnt,
                                                 int* __restrict__ off,
                                                 int* __restrict__ cur) {
  __shared__ int part[SCAN_T];
  const int t = threadIdx.x;
  const int base = t * SCAN_ITEMS;
  int loc[SCAN_ITEMS];
  int sum = 0;
#pragma unroll
  for (int j = 0; j < SCAN_ITEMS; ++j) {
    int idx = base + j;
    int c = (idx < N_NODES) ? cnt[idx] : 0;
    loc[j] = sum;              // thread-local exclusive prefix
    sum += c;
  }
  part[t] = sum;
  __syncthreads();
  for (int d = 1; d < SCAN_T; d <<= 1) {   // Hillis-Steele inclusive scan
    int v = (t >= d) ? part[t - d] : 0;
    __syncthreads();
    part[t] += v;
    __syncthreads();
  }
  int ex = (t == 0) ? 0 : part[t - 1];
#pragma unroll
  for (int j = 0; j < SCAN_ITEMS; ++j) {
    int idx = base + j;
    if (idx < N_NODES) { int o = ex + loc[j]; off[idx] = o; cur[idx] = o; }
  }
}

__global__ void k_sort(const int* __restrict__ ei, int* __restrict__ cur,
                       int* __restrict__ sorted) {
  int e = blockIdx.x * blockDim.x + threadIdx.x;
  if (e >= N_EDGES) return;
  int d = ei[N_EDGES + e];
  int pos = atomicAdd(&cur[d], 1);
  sorted[pos] = e;
}

// ---- fused per-node online-softmax attention + aggregation (no atomics) ----
// wave per destination node; x_dst row and (max,sum,acc) live in registers;
// each incoming x_src row is streamed once from L2.
__global__ void k_node(const int* __restrict__ ei, const int* __restrict__ sorted,
                       const int* __restrict__ off, const int* __restrict__ cnt,
                       const float* __restrict__ xsrc, const float* __restrict__ xdst,
                       float* __restrict__ z) {
  const int wid  = (blockIdx.x * blockDim.x + threadIdx.x) >> 5;  // node id
  const int lane = threadIdx.x & 31;
  if (wid >= N_NODES) return;
  const float4* pd = (const float4*)(xdst + (size_t)wid * DIM);
  float4 hd[4];
#pragma unroll
  for (int t = 0; t < 4; ++t) hd[t] = pd[lane + 32 * t];

  float4 acc[4];
#pragma unroll
  for (int t = 0; t < 4; ++t) acc[t] = make_float4(0.f, 0.f, 0.f, 0.f);
  float m = -__builtin_inff();
  float ssum = 0.0f;

  const int start = off[wid];
  const int end   = start + cnt[wid];
  for (int j = start; j < end; ++j) {
    const int e = sorted[j];
    const int s = ei[e];
    const float4* ps = (const float4*)(xsrc + (size_t)s * DIM);
    float4 hs[4];
    float dot = 0.0f;
#pragma unroll
    for (int t = 0; t < 4; ++t) {
      hs[t] = ps[lane + 32 * t];
      dot += hs[t].x * hd[t].x + hs[t].y * hd[t].y +
             hs[t].z * hd[t].z + hs[t].w * hd[t].w;
    }
#pragma unroll
    for (int o = 16; o > 0; o >>= 1) dot += __shfl_xor(dot, o, 32);
    const float v  = dot * 0.04419417382415922f;       // 1/sqrt(512)
    const float nm = fmaxf(m, v);
    const float sc = __expf(m - nm);                   // first iter: exp(-inf)=0
    const float w  = __expf(v - nm);
    ssum = ssum * sc + w;
#pragma unroll
    for (int t = 0; t < 4; ++t) {
      acc[t].x = acc[t].x * sc + w * hs[t].x;
      acc[t].y = acc[t].y * sc + w * hs[t].y;
      acc[t].z = acc[t].z * sc + w * hs[t].z;
      acc[t].w = acc[t].w * sc + w * hs[t].w;
    }
    m = nm;
  }
  const float inv = 1.0f / (ssum + 1e-16f);            // matches ex/(sum+1e-16)
  float4* zr = (float4*)(z + (size_t)wid * DIM);
#pragma unroll
  for (int t = 0; t < 4; ++t) {
    zr[lane + 32 * t] = make_float4(acc[t].x * inv, acc[t].y * inv,
                                    acc[t].z * inv, acc[t].w * inv);
  }
}

// ---- prediction head: out = (z @ W_pred + b_pred) * (mask==1) --------------
__global__ void k_pred(const float* __restrict__ z, const float* __restrict__ Wp,
                       const float* __restrict__ bp, const int* __restrict__ mask,
                       float* __restrict__ out) {
  const int wid  = (blockIdx.x * blockDim.x + threadIdx.x) >> 5;  // node id
  const int lane = threadIdx.x & 31;
  if (wid >= N_NODES) return;
  const float* zr = z + (size_t)wid * DIM;
  float a0 = 0.0f, a1 = 0.0f;
#pragma unroll
  for (int t = 0; t < 16; ++t) {
    int k = lane + 32 * t;
    float v = zr[k];
    a0 += v * Wp[2 * k];
    a1 += v * Wp[2 * k + 1];
  }
#pragma unroll
  for (int o = 16; o > 0; o >>= 1) {
    a0 += __shfl_xor(a0, o, 32);
    a1 += __shfl_xor(a1, o, 32);
  }
  if (lane == 0) {
    float mm = (mask[wid] == 1) ? 1.0f : 0.0f;
    out[2 * wid]     = (a0 + bp[0]) * mm;
    out[2 * wid + 1] = (a1 + bp[1]) * mm;
  }
}

extern "C" void kernel_launch(void* const* d_in, const int* in_sizes, int n_in,
                              void* d_out, int out_size, void* d_ws, size_t ws_size,
                              hipStream_t stream) {
  (void)in_sizes; (void)n_in; (void)out_size; (void)ws_size;
  // setup_inputs() order: x, edge_index, tg_mask, W_src, b_src, W_dst, b_dst, W_pred, b_pred
  const float* x     = (const float*)d_in[0];
  const int*   ei    = (const int*)  d_in[1];
  const int*   tgm   = (const int*)  d_in[2];
  const float* W_src = (const float*)d_in[3];
  const float* b_src = (const float*)d_in[4];
  const float* W_dst = (const float*)d_in[5];
  const float* b_dst = (const float*)d_in[6];
  const float* W_prd = (const float*)d_in[7];
  const float* b_prd = (const float*)d_in[8];
  float* out = (float*)d_out;

  // workspace carve-out (~73.4 MB total)
  uint8_t* ws = (uint8_t*)d_ws;
  uint32_t* xbf  = (uint32_t*)ws;  ws += (size_t)N_NODES * KPAIRS * 4;   // 10.24 MB
  uint32_t* Wspk = (uint32_t*)ws;  ws += (size_t)DIM * KPAIRS * 4;       // 0.5 MB
  uint32_t* Wdpk = (uint32_t*)ws;  ws += (size_t)DIM * KPAIRS * 4;       // 0.5 MB
  float* xsrc    = (float*)ws;     ws += (size_t)N_NODES * DIM * 4;      // 20.48 MB
  float* xdst    = (float*)ws;     ws += (size_t)N_NODES * DIM * 4;      // 20.48 MB
  float* z       = (float*)ws;     ws += (size_t)N_NODES * DIM * 4;      // 20.48 MB
  int*   cnt     = (int*)ws;       ws += (size_t)N_NODES * 4;
  int*   off     = (int*)ws;       ws += (size_t)N_NODES * 4;
  int*   cur     = (int*)ws;       ws += (size_t)N_NODES * 4;
  int*   sorted  = (int*)ws;       ws += (size_t)N_EDGES * 4;            // 0.64 MB

  const int T = 256;
  // 1) convert / pack for WMMA
  k_cvt_x <<<(N_NODES * KPAIRS + T - 1) / T, T, 0, stream>>>(x, xbf);
  k_pack_w<<<(DIM * KPAIRS + T - 1) / T, T, 0, stream>>>(W_src, Wspk);
  k_pack_w<<<(DIM * KPAIRS + T - 1) / T, T, 0, stream>>>(W_dst, Wdpk);
  // 2) dual projection GEMM via WMMA (20000 16x16 tiles, 8 waves/block)
  k_gemm_wmma<<<(N_NODES / 16) * (DIM / 16) / 8, T, 0, stream>>>(
      xbf, Wspk, Wdpk, b_src, b_dst, xsrc, xdst);
  // 3) counting sort of edges by destination (int atomics only)
  k_zero_cnt<<<(N_NODES + T - 1) / T, T, 0, stream>>>(cnt);
  k_hist    <<<(N_EDGES + T - 1) / T, T, 0, stream>>>(ei, cnt);
  k_scan    <<<1, SCAN_T, 0, stream>>>(cnt, off, cur);
  k_sort    <<<(N_EDGES + T - 1) / T, T, 0, stream>>>(ei, cur, sorted);
  // 4) fused score + online softmax + aggregate, wave per node, no atomics
  k_node<<<(N_NODES * 32 + T - 1) / T, T, 0, stream>>>(
      ei, sorted, off, cnt, xsrc, xdst, z);
  // 5) masked prediction head (wave per node)
  k_pred<<<(N_NODES * 32 + T - 1) / T, T, 0, stream>>>(z, W_prd, b_prd, tgm, out);
}
